// Gagnn_22582938042571
// MI455X (gfx1250) — matrix operand order repaired and checked by reading, not compiled, not run on hardware
//
#include <hip/hip_runtime.h>

typedef __attribute__((ext_vector_type(16))) _Float16 v16h;
typedef __attribute__((ext_vector_type(8)))  _Float16 v8h;
typedef __attribute__((ext_vector_type(8)))  float    v8f;

#define BQ 16
#define CITYN 1000
#define NGROUP 15
#define SEQL 24
#define NINPUT 8
#define NFEAT 12
#define XEMD 64
#define LOCEMD 16
#define EDGEHD 32
#define GNNHD 128
#define PREDD 12
#define NEDGE 16000
#define NPAIR 210

__device__ __forceinline__ v8f wmma16(v16h a, v16h b, v8f c) {
  return __builtin_amdgcn_wmma_f32_16x16x32_f16(false, a, false, b, (short)0, c, false, false);
}
// Branch-free activations: native v_exp_f32 + fast divide (no libm exec-branches).
__device__ __forceinline__ float sigm(float x) {
  return __fdividef(1.f, 1.f + __expf(-x));
}
__device__ __forceinline__ float tanh_fast(float x) {
  // 1 - 2/(e^{2x}+1); exp->inf gives 1, exp->0 gives -1 (graceful saturation)
  return 1.f - __fdividef(2.f, __expf(2.f * x) + 1.f);
}

// ---------------------------------------------------------------------------
// Weight swizzle: pack W[K,N] (f32 row-major) into WMMA B-register order so a
// wave fills its B operand with one contiguous 32B load per (chunk, ntile):
//   out[((c*nTiles + nt)*32 + lane)*16 + e] = f16( W[(c*32 + (lane>=16?8:0) + (e<8?e:e+8))*N
//                                                   + nt*16 + (lane&15)] )
// ---------------------------------------------------------------------------
__global__ void swizzle_w_kernel(const float* __restrict__ W, int K, int N,
                                 _Float16* __restrict__ out, int nTiles, int tot) {
  int idx = blockIdx.x * blockDim.x + threadIdx.x;
  if (idx >= tot) return;
  int tile = idx >> 9, l = (idx >> 4) & 31, e = idx & 15;
  int c = tile / nTiles, nt = tile % nTiles;
  int n = nt * 16 + (l & 15);
  int k = c * 32 + ((l >> 4) ? 8 : 0) + (e < 8 ? e : e + 8);
  float v = (k < K && n < N) ? W[(size_t)k * N + n] : 0.f;
  out[idx] = (_Float16)v;
}

// ---------------------------------------------------------------------------
// Fused GEMM (templated on source widths + N): out = act(concat(S1,S2,S3) @ W + b)
// A operand: per-element pointer/index *selects* (no exec branches) + one
// unconditional in-bounds load. B operand: one v16h load from the swizzled W.
// Epilogue: direct store or f32 atomic scatter-add (fused segment-sum).
// All M are multiples of 16 -> no row guards.
// ---------------------------------------------------------------------------
template <int D1, int D2, int D3, int NN>
__global__ __launch_bounds__(256) void fused_gemm_t(
    const float* __restrict__ p1, const int* __restrict__ g1, const float* __restrict__ sc1,
    const float* __restrict__ p2, const int* __restrict__ g2, const float* __restrict__ sc2,
    const float* __restrict__ p3, const int* __restrict__ g3, const float* __restrict__ sc3,
    const _Float16* __restrict__ Wsw, const float* __restrict__ bias,
    int M, int act, float* __restrict__ out, const int* __restrict__ scat)
{
  constexpr int K  = D1 + D2 + D3;
  constexpr int NC = (K + 31) / 32;
  constexpr int NT = (NN + 15) / 16;

  const int lane = threadIdx.x & 31;
  const int wv   = threadIdx.x >> 5;
  const int mt   = blockIdx.x * 8 + wv;
  if (mt >= ((M + 15) >> 4)) return;       // wave-uniform
  const int m = lane & 15, hi = lane >> 4;
  const int row0 = mt << 4;
  const int rowA = row0 + m;               // M % 16 == 0 -> always < M

  const int r1 = g1 ? g1[rowA] : rowA;
  const float s1v = sc1 ? sc1[r1] : 1.f;
  int r2 = 0; float s2v = 1.f;
  if constexpr (D2 > 0) { r2 = g2 ? g2[rowA] : rowA; if (sc2) s2v = sc2[r2]; }
  int r3 = 0; float s3v = 1.f;
  if constexpr (D3 > 0) { r3 = g3 ? g3[rowA] : rowA; if (sc3) s3v = sc3[r3]; }

  v16h A[NC];
  #pragma unroll
  for (int c = 0; c < NC; ++c) {
    #pragma unroll
    for (int e = 0; e < 16; ++e) {
      const int kb = (c << 5) + (e < 8 ? e : e + 8);  // compile-time part
      const int k  = kb + (hi ? 8 : 0);               // lane-half dependent
      const float* ap = p1; int ix = 0; float sv = 0.f;
      if (k < D1)                 { ap = p1; ix = r1 * D1 + k;            sv = s1v; }
      else if (k < D1 + D2)       { ap = p2; ix = r2 * D2 + (k - D1);     sv = s2v; }
      else if (k < K)             { ap = p3; ix = r3 * D3 + (k - D1 - D2); sv = s3v; }
      A[c][e] = (_Float16)(ap[ix] * sv);              // select + one load, no branch
    }
  }

  #pragma unroll
  for (int nt = 0; nt < NT; ++nt) {
    const int nb = (nt << 4) + m;
    v8f acc = {};
    #pragma unroll
    for (int c = 0; c < NC; ++c) {
      const v16h Bv = *(const v16h*)&Wsw[(size_t)(((c * NT + nt) * 32 + lane)) * 16];
      acc = wmma16(A[c], Bv, acc);
    }
    const bool nok = (NN % 16 == 0) ? true : (nb < NN);
    if (nok) {
      const float bv = bias[nb];
      #pragma unroll
      for (int r = 0; r < 8; ++r) {
        const int rr = row0 + r + (hi << 3);
        float v = acc[r] + bv;
        if (act) v = fmaxf(v, 0.f);
        if (scat) atomicAdd(&out[(size_t)scat[rr] * NN + nb], v);
        else      out[(size_t)rr * NN + nb] = v;
      }
    }
  }
}

template <int D1, int D2, int D3, int NN>
static inline void gemm_launch(const float* p1, const int* g1, const float* sc1,
                               const float* p2, const int* g2, const float* sc2,
                               const float* p3, const int* g3, const float* sc3,
                               const _Float16* Wsw, const float* bias, int M, int act,
                               float* out, const int* scat, hipStream_t s) {
  int blocks = (((M + 15) >> 4) + 7) / 8;
  fused_gemm_t<D1, D2, D3, NN><<<blocks, 256, 0, s>>>(p1, g1, sc1, p2, g2, sc2,
                                                      p3, g3, sc3, Wsw, bias, M, act, out, scat);
}

// ---------------------------------------------------------------------------
// LSTM: one wave = 16 sequences. Weights swizzled into B-register order in LDS
// at block init (one v16h LDS load per WMMA; compiler keeps them resident in
// VGPRs across timesteps). Recurrent h round-trips through a per-wave LDS stage
// read back as contiguous v8h (16B) runs. 48 WMMAs / step.
// Combined K rows: 0-7 = Wih, 8-31 = 0, 32-95 = Whh.  N tiles: nt = gate*4 + u.
// ---------------------------------------------------------------------------
__global__ __launch_bounds__(128) void lstm_kernel(
    const float* __restrict__ x, const float* __restrict__ Wih,
    const float* __restrict__ Whh, const float* __restrict__ bias,
    float* __restrict__ xe)
{
  __shared__ _Float16 Wls[48 * 512];      // 48 tiles * 32 lanes * 16 elems (49152B)
  __shared__ float bl[256];
  __shared__ _Float16 hst[4][16 * 64];

  for (int idx = threadIdx.x; idx < 48 * 512; idx += blockDim.x) {
    int tile = idx >> 9, l = (idx >> 4) & 31, e = idx & 15;
    int ch = tile >> 4, nt = tile & 15;
    int n = nt * 16 + (l & 15);
    int k = ch * 32 + ((l >> 4) ? 8 : 0) + (e < 8 ? e : e + 8);
    float v = 0.f;
    if (k < 8) v = Wih[k * 256 + n];
    else if (k >= 32 && k < 96) v = Whh[(k - 32) * 256 + n];
    Wls[idx] = (_Float16)v;
  }
  for (int idx = threadIdx.x; idx < 256; idx += blockDim.x) bl[idx] = bias[idx];
  __syncthreads();

  const int lane = threadIdx.x & 31;
  const int wv = threadIdx.x >> 5;
  const int m = lane & 15, hi = lane >> 4;
  const int seq0 = blockIdx.x * 64 + wv * 16;
  const int row = seq0 + m;

  for (int idx = lane; idx < 16 * 64; idx += 32) hst[wv][idx] = (_Float16)0.f;
  v8f carr[4] = {};

  for (int t = 0; t < SEQL; ++t) {
    // x_t: 8 contiguous, always-in-bounds loads (t*12+7 < 288); zero for hi lanes
    float xv[8];
    #pragma unroll
    for (int j = 0; j < 8; ++j) xv[j] = x[(size_t)row * (SEQL * NFEAT) + t * NFEAT + j];
    v16h A0;
    #pragma unroll
    for (int e = 0; e < 16; ++e)
      A0[e] = (!hi && e < 8) ? (_Float16)xv[e] : (_Float16)0.f;

    // h (A-layout): two contiguous 16B runs per chunk from the LDS stage
    const v8h* hrow = (const v8h*)&hst[wv][m * 64];
    v8h a1lo = hrow[hi],     a1hi = hrow[2 + hi];
    v8h a2lo = hrow[4 + hi], a2hi = hrow[6 + hi];
    v16h A1, A2;
    #pragma unroll
    for (int e = 0; e < 8; ++e) {
      A1[e] = a1lo[e]; A1[8 + e] = a1hi[e];
      A2[e] = a2lo[e]; A2[8 + e] = a2hi[e];
    }

    #pragma unroll
    for (int u = 0; u < 4; ++u) {         // 16-col slices of the 64-wide state
      v8f z[4];
      #pragma unroll
      for (int gate = 0; gate < 4; ++gate) {  // i, f, g, o
        const int nt = gate * 4 + u;
        v8f acc = {};
        #pragma unroll
        for (int ch = 0; ch < 3; ++ch) {
          const v16h Bv = *(const v16h*)&Wls[((ch * 16 + nt) * 32 + lane) * 16];
          acc = wmma16(ch == 0 ? A0 : (ch == 1 ? A1 : A2), Bv, acc);
        }
        z[gate] = acc;
      }
      const int colb = u * 16 + m;
      #pragma unroll
      for (int r = 0; r < 8; ++r) {
        float zi = z[0][r] + bl[colb];
        float zf = z[1][r] + bl[64 + colb];
        float zg = z[2][r] + bl[128 + colb];
        float zo = z[3][r] + bl[192 + colb];
        float cv = sigm(zf) * carr[u][r] + sigm(zi) * tanh_fast(zg);
        carr[u][r] = cv;
        float hv = sigm(zo) * tanh_fast(cv);
        hst[wv][(r + hi * 8) * 64 + u * 16 + m] = (_Float16)hv;
      }
    }
  }
  for (int idx = lane; idx < 16 * 64; idx += 32) {
    int mm = idx >> 6, col = idx & 63;
    xe[((size_t)(seq0 + mm)) * XEMD + col] = (float)hst[wv][idx];
  }
}

// ------------------------------ small kernels ------------------------------
__global__ void zero_f32_kernel(float* p, int n) {
  int i = blockIdx.x * blockDim.x + threadIdx.x;
  if (i < n) p[i] = 0.f;
}
__global__ void fill_f32_kernel(float* p, float v, int n) {
  int i = blockIdx.x * blockDim.x + threadIdx.x;
  if (i < n) p[i] = v;
}
__global__ void softmax_kernel(const float* __restrict__ w, float* __restrict__ o) {
  int c = blockIdx.x * blockDim.x + threadIdx.x;
  if (c >= CITYN) return;
  float mx = -1e30f;
  for (int g = 0; g < NGROUP; ++g) mx = fmaxf(mx, w[c * NGROUP + g]);
  float s = 0.f;
  for (int g = 0; g < NGROUP; ++g) s += __expf(w[c * NGROUP + g] - mx);
  float inv = __fdividef(1.f, s);
  for (int g = 0; g < NGROUP; ++g) o[c * NGROUP + g] = __expf(w[c * NGROUP + g] - mx) * inv;
}
__global__ void locembed_kernel(const float* __restrict__ loc, const float* __restrict__ W,
                                const float* __restrict__ b, float* __restrict__ o) {
  int i = blockIdx.x * blockDim.x + threadIdx.x;
  if (i >= CITYN * LOCEMD) return;
  int c = i / LOCEMD, d = i % LOCEMD;
  o[i] = loc[c * 2] * W[d] + loc[c * 2 + 1] * W[LOCEMD + d] + b[d];
}
__global__ void uem_kernel(const float* __restrict__ x, const float* __restrict__ u1,
                           const float* __restrict__ u2, const float* __restrict__ u3,
                           float* __restrict__ o) {
  int b = blockIdx.x * blockDim.x + threadIdx.x;
  if (b >= BQ) return;
  const float* xb = x + (((size_t)b * CITYN) * SEQL + (SEQL - 1)) * NFEAT;
  int mo = (int)(xb[8] + 0.5f);
  int wd = (int)(xb[10] + 0.5f);
  int hr = (int)(xb[11] + 0.5f);
  for (int j = 0; j < 8; ++j) {
    o[b * 24 + j]      = u1[mo * 8 + j];
    o[b * 24 + 8 + j]  = u2[wd * 8 + j];
    o[b * 24 + 16 + j] = u3[hr * 8 + j];
  }
}
// out[b,g,d] = sum_c w[c,g] * (d<dA ? A[b,c,d] : Bp[c,d-dA])
__global__ void grouping_kernel(const float* __restrict__ w, const float* __restrict__ A, int dA,
                                const float* __restrict__ Bp, int dB, float* __restrict__ out) {
  int D = dA + dB;
  int t = blockIdx.x * blockDim.x + threadIdx.x;
  if (t >= BQ * NGROUP * D) return;
  int d = t % D, g = (t / D) % NGROUP, b = t / (D * NGROUP);
  float acc = 0.f;
  for (int c = 0; c < CITYN; ++c) {
    float v = (d < dA) ? A[((size_t)(b * CITYN + c)) * dA + d] : Bp[c * dB + (d - dA)];
    acc += w[c * NGROUP + g] * v;
  }
  out[((size_t)(b * NGROUP + g)) * D + d] = acc;
}
// out[b,c,d] = sum_g w[c,g] * gx[b,g,d]   (D = GNNHD)
__global__ void ungroup_kernel(const float* __restrict__ w, const float* __restrict__ gx,
                               float* __restrict__ out) {
  int t = blockIdx.x * blockDim.x + threadIdx.x;
  if (t >= BQ * CITYN * GNNHD) return;
  int d = t % GNNHD, c = (t / GNNHD) % CITYN, b = t / (GNNHD * CITYN);
  float acc = 0.f;
  for (int g = 0; g < NGROUP; ++g)
    acc += w[c * NGROUP + g] * gx[((size_t)(b * NGROUP + g)) * GNNHD + d];
  out[t] = acc;
}
__global__ void group_edges_kernel(int* grow, int* gcol, int* gb) {
  int t = blockIdx.x * blockDim.x + threadIdx.x;
  if (t >= BQ * NPAIR) return;
  int b = t / NPAIR, p = t % NPAIR;
  int i = p / 14, jj = p % 14;
  int j = jj + (jj >= i ? 1 : 0);
  grow[t] = b * NGROUP + i;
  gcol[t] = b * NGROUP + j;
  gb[t] = b;
}
__global__ void global_edges_kernel(const int* __restrict__ ei, int* grow, int* gcol, int* eg) {
  int t = blockIdx.x * blockDim.x + threadIdx.x;
  if (t >= BQ * NEDGE) return;
  int b = t / NEDGE, e = t % NEDGE;
  grow[t] = b * CITYN + ei[e];
  gcol[t] = b * CITYN + ei[NEDGE + e];
  eg[t] = e;
}
__global__ void count_kernel(const int* __restrict__ ei, float* cnt) {
  int e = blockIdx.x * blockDim.x + threadIdx.x;
  if (e < NEDGE) atomicAdd(&cnt[ei[NEDGE + e]], 1.f);
}
__global__ void inv_kernel(const float* __restrict__ cnt, float* invg) {
  int i = blockIdx.x * blockDim.x + threadIdx.x;
  if (i < BQ * CITYN) invg[i] = __fdividef(1.f, fmaxf(cnt[i % CITYN], 1.f));
}

// ---------------------------------------------------------------------------
extern "C" void kernel_launch(void* const* d_in, const int* in_sizes, int n_in,
                              void* d_out, int out_size, void* d_ws, size_t ws_size,
                              hipStream_t stream) {
  (void)in_sizes; (void)n_in; (void)out_size; (void)ws_size;
  // d_in order: jax pytree flatten (keys sorted alphabetically at every level).
  const int*   edge_index = (const int*)d_in[0];
  const float* edge_w = (const float*)d_in[1];
  const float* loc    = (const float*)d_in[2];
  const float *dgg0m1W=(const float*)d_in[3],  *dgg0m1b=(const float*)d_in[4];
  const float *dgg0m2W=(const float*)d_in[5],  *dgg0m2b=(const float*)d_in[6];
  const float *dgg1m1W=(const float*)d_in[7],  *dgg1m1b=(const float*)d_in[8];
  const float *dgg1m2W=(const float*)d_in[9],  *dgg1m2b=(const float*)d_in[10];
  const float *dgr0m1W=(const float*)d_in[11], *dgr0m1b=(const float*)d_in[12];
  const float *dgr0m2W=(const float*)d_in[13], *dgr0m2b=(const float*)d_in[14];
  const float *dgr1m1W=(const float*)d_in[15], *dgr1m1b=(const float*)d_in[16];
  const float *dgr1m2W=(const float*)d_in[17], *dgr1m2b=(const float*)d_in[18];
  const float *dxeW=(const float*)d_in[19],    *dxeb=(const float*)d_in[20];
  const float *einfW=(const float*)d_in[21],   *einfb=(const float*)d_in[22];
  const float *gg0m1W=(const float*)d_in[23],  *gg0m1b=(const float*)d_in[24];
  const float *gg0m2W=(const float*)d_in[25],  *gg0m2b=(const float*)d_in[26];
  const float *gg1m1W=(const float*)d_in[27],  *gg1m1b=(const float*)d_in[28];
  const float *gg1m2W=(const float*)d_in[29],  *gg1m2b=(const float*)d_in[30];
  const float *gr0m1W=(const float*)d_in[31],  *gr0m1b=(const float*)d_in[32];
  const float *gr0m2W=(const float*)d_in[33],  *gr0m2b=(const float*)d_in[34];
  const float *gr1m1W=(const float*)d_in[35],  *gr1m1b=(const float*)d_in[36];
  const float *gr1m2W=(const float*)d_in[37],  *gr1m2b=(const float*)d_in[38];
  const float *locW=(const float*)d_in[39],    *locb=(const float*)d_in[40];
  const float *Whh=(const float*)d_in[41];
  const float *Wih=(const float*)d_in[42];
  const float *lstmb=(const float*)d_in[43];
  const float *p1W=(const float*)d_in[44],     *p1b=(const float*)d_in[45];
  const float *p2W=(const float*)d_in[46],     *p2b=(const float*)d_in[47];
  const float *u1=(const float*)d_in[48];
  const float *u2=(const float*)d_in[49];
  const float *u3=(const float*)d_in[50];
  const float *wraw=(const float*)d_in[51];
  const float *x=(const float*)d_in[52];

  float* F = (float*)d_ws;
  size_t off = 0;
  auto take = [&](size_t n) { size_t c = off; off += (n + 63) & ~(size_t)63; return c; };
  float* wsm   = F + take(CITYN * NGROUP);
  float* loce  = F + take(CITYN * LOCEMD);
  float* xe    = F + take((size_t)BQ * CITYN * XEMD);
  float* uem   = F + take(BQ * 24);
  float* gx0   = F + take(BQ * NGROUP * (XEMD + LOCEMD));
  float* gew   = F + take((size_t)BQ * NPAIR * EDGEHD);
  float* gx1   = F + take(BQ * NGROUP * GNNHD);
  float* gx2   = F + take(BQ * NGROUP * GNNHD);
  float* sg    = F + take(BQ * NGROUP * GNNHD);
  float* inv14 = F + take(BQ * NGROUP);
  float* cnt   = F + take(CITYN);
  float* invg  = F + take(BQ * CITYN);
  float* sbuf  = F + take((size_t)BQ * CITYN * GNNHD);
  float* ung1  = F + take((size_t)BQ * CITYN * GNNHD);
  float* nx1   = F + take((size_t)BQ * CITYN * GNNHD);
  float* nx2   = F + take((size_t)BQ * CITYN * GNNHD);
  float* xd    = F + take((size_t)BQ * CITYN * XEMD);
  float* gdx0  = F + take(BQ * NGROUP * XEMD);
  float* gdx1  = F + take(BQ * NGROUP * GNNHD);
  float* gdx2  = F + take(BQ * NGROUP * GNNHD);
  float* ung2  = F + take((size_t)BQ * CITYN * GNNHD);
  float* nd1   = F + take((size_t)BQ * CITYN * GNNHD);
  float* nd2   = F + take((size_t)BQ * CITYN * GNNHD);
  float* pbuf  = F + take((size_t)BQ * CITYN * 16);
  int* growS = (int*)(F + take(BQ * NPAIR));
  int* gcolS = (int*)(F + take(BQ * NPAIR));
  int* gbS   = (int*)(F + take(BQ * NPAIR));
  int* growG = (int*)(F + take((size_t)BQ * NEDGE));
  int* gcolG = (int*)(F + take((size_t)BQ * NEDGE));
  int* egath = (int*)(F + take((size_t)BQ * NEDGE));
  _Float16* swBase = (_Float16*)(F + take(256 * 1024));  // 1MB region for swizzled weights

  auto g1d = [](int n) { return (n + 255) / 256; };
  auto zero = [&](float* p, int n) { zero_f32_kernel<<<g1d(n), 256, 0, stream>>>(p, n); };
  size_t swOff = 0;
  auto swizzle = [&](const float* W, int K, int N) -> const _Float16* {
    int nc = (K + 31) / 32, nt = (N + 15) / 16;
    int tot = nc * nt * 512;
    _Float16* dst = swBase + swOff;
    swOff += (size_t)tot;
    swizzle_w_kernel<<<g1d(tot), 256, 0, stream>>>(W, K, N, dst, nt, tot);
    return dst;
  };

  // -------- swizzle all weight matrices into WMMA B layout --------
  const _Float16* sw_einf  = swizzle(einfW, 184, 32);
  const _Float16* sw_gr0m1 = swizzle(gr0m1W, 112, 128);
  const _Float16* sw_gr0m2 = swizzle(gr0m2W, 208, 128);
  const _Float16* sw_gr1m1 = swizzle(gr1m1W, 160, 128);
  const _Float16* sw_gr1m2 = swizzle(gr1m2W, 256, 128);
  const _Float16* sw_gg0m1 = swizzle(gg0m1W, 193, 128);
  const _Float16* sw_gg0m2 = swizzle(gg0m2W, 320, 128);
  const _Float16* sw_gg1m1 = swizzle(gg1m1W, 129, 128);
  const _Float16* sw_gg1m2 = swizzle(gg1m2W, 256, 128);
  const _Float16* sw_dxe   = swizzle(dxeW, 128, 64);
  const _Float16* sw_dgr0m1= swizzle(dgr0m1W, 96, 128);
  const _Float16* sw_dgr0m2= swizzle(dgr0m2W, 192, 128);
  const _Float16* sw_dgr1m1= swizzle(dgr1m1W, 160, 128);
  const _Float16* sw_dgr1m2= swizzle(dgr1m2W, 256, 128);
  const _Float16* sw_dgg0m1= swizzle(dgg0m1W, 193, 128);
  const _Float16* sw_dgg0m2= swizzle(dgg0m2W, 320, 128);
  const _Float16* sw_dgg1m1= swizzle(dgg1m1W, 129, 128);
  const _Float16* sw_dgg1m2= swizzle(dgg1m2W, 256, 128);
  const _Float16* sw_p1    = swizzle(p1W, 128, 16);
  const _Float16* sw_p2    = swizzle(p2W, 16, 12);

  // -------- setup --------
  zero(cnt, CITYN);
  count_kernel<<<g1d(NEDGE), 256, 0, stream>>>(edge_index, cnt);
  inv_kernel<<<g1d(BQ * CITYN), 256, 0, stream>>>(cnt, invg);
  fill_f32_kernel<<<1, 256, 0, stream>>>(inv14, 1.f / 14.f, BQ * NGROUP);
  group_edges_kernel<<<g1d(BQ * NPAIR), 256, 0, stream>>>(growS, gcolS, gbS);
  global_edges_kernel<<<g1d(BQ * NEDGE), 256, 0, stream>>>(edge_index, growG, gcolG, egath);
  softmax_kernel<<<g1d(CITYN), 256, 0, stream>>>(wraw, wsm);
  locembed_kernel<<<g1d(CITYN * LOCEMD), 256, 0, stream>>>(loc, locW, locb, loce);
  uem_kernel<<<1, 32, 0, stream>>>(x, u1, u2, u3, uem);

  // -------- encoder --------
  lstm_kernel<<<BQ * CITYN / 64, 128, 0, stream>>>(x, Wih, Whh, lstmb, xe);
  grouping_kernel<<<g1d(BQ * NGROUP * 80), 256, 0, stream>>>(wsm, xe, XEMD, loce, LOCEMD, gx0);
  gemm_launch<80, 80, 24, 32>(gx0, growS, nullptr, gx0, gcolS, nullptr, uem, gbS, nullptr,
                              sw_einf, einfb, BQ * NPAIR, 1, gew, nullptr, stream);
  zero(sg, BQ * NGROUP * GNNHD);
  gemm_launch<80, 32, 0, 128>(gx0, growS, nullptr, gew, nullptr, nullptr, nullptr, nullptr, nullptr,
                              sw_gr0m1, gr0m1b, BQ * NPAIR, 1, sg, gcolS, stream);
  gemm_launch<80, 128, 0, 128>(gx0, nullptr, nullptr, sg, nullptr, inv14, nullptr, nullptr, nullptr,
                               sw_gr0m2, gr0m2b, BQ * NGROUP, 1, gx1, nullptr, stream);
  zero(sg, BQ * NGROUP * GNNHD);
  gemm_launch<128, 32, 0, 128>(gx1, growS, nullptr, gew, nullptr, nullptr, nullptr, nullptr, nullptr,
                               sw_gr1m1, gr1m1b, BQ * NPAIR, 1, sg, gcolS, stream);
  gemm_launch<128, 128, 0, 128>(gx1, nullptr, nullptr, sg, nullptr, inv14, nullptr, nullptr, nullptr,
                                sw_gr1m2, gr1m2b, BQ * NGROUP, 1, gx2, nullptr, stream);
  ungroup_kernel<<<g1d(BQ * CITYN * GNNHD), 256, 0, stream>>>(wsm, gx2, ung1);
  zero(sbuf, BQ * CITYN * GNNHD);
  gemm_launch<64, 128, 1, 128>(xe, growG, nullptr, ung1, growG, nullptr, edge_w, egath, nullptr,
                               sw_gg0m1, gg0m1b, BQ * NEDGE, 1, sbuf, gcolG, stream);
  gemm_launch<64, 128, 128, 128>(xe, nullptr, nullptr, ung1, nullptr, nullptr, sbuf, nullptr, invg,
                                 sw_gg0m2, gg0m2b, BQ * CITYN, 1, nx1, nullptr, stream);
  zero(sbuf, BQ * CITYN * GNNHD);
  gemm_launch<128, 1, 0, 128>(nx1, growG, nullptr, edge_w, egath, nullptr, nullptr, nullptr, nullptr,
                              sw_gg1m1, gg1m1b, BQ * NEDGE, 1, sbuf, gcolG, stream);
  gemm_launch<128, 128, 0, 128>(nx1, nullptr, nullptr, sbuf, nullptr, invg, nullptr, nullptr, nullptr,
                                sw_gg1m2, gg1m2b, BQ * CITYN, 1, nx2, nullptr, stream);

  // -------- decoder (uses RAW w, not softmax) --------
  gemm_launch<128, 0, 0, 64>(nx2, nullptr, nullptr, nullptr, nullptr, nullptr, nullptr, nullptr, nullptr,
                             sw_dxe, dxeb, BQ * CITYN, 0, xd, nullptr, stream);
  grouping_kernel<<<g1d(BQ * NGROUP * XEMD), 256, 0, stream>>>(wraw, xd, XEMD, nullptr, 0, gdx0);
  zero(sg, BQ * NGROUP * GNNHD);
  gemm_launch<64, 32, 0, 128>(gdx0, growS, nullptr, gew, nullptr, nullptr, nullptr, nullptr, nullptr,
                              sw_dgr0m1, dgr0m1b, BQ * NPAIR, 1, sg, gcolS, stream);
  gemm_launch<64, 128, 0, 128>(gdx0, nullptr, nullptr, sg, nullptr, inv14, nullptr, nullptr, nullptr,
                               sw_dgr0m2, dgr0m2b, BQ * NGROUP, 1, gdx1, nullptr, stream);
  zero(sg, BQ * NGROUP * GNNHD);
  gemm_launch<128, 32, 0, 128>(gdx1, growS, nullptr, gew, nullptr, nullptr, nullptr, nullptr, nullptr,
                               sw_dgr1m1, dgr1m1b, BQ * NPAIR, 1, sg, gcolS, stream);
  gemm_launch<128, 128, 0, 128>(gdx1, nullptr, nullptr, sg, nullptr, inv14, nullptr, nullptr, nullptr,
                                sw_dgr1m2, dgr1m2b, BQ * NGROUP, 1, gdx2, nullptr, stream);
  ungroup_kernel<<<g1d(BQ * CITYN * GNNHD), 256, 0, stream>>>(wraw, gdx2, ung2);
  zero(sbuf, BQ * CITYN * GNNHD);
  gemm_launch<64, 128, 1, 128>(xd, growG, nullptr, ung2, growG, nullptr, edge_w, egath, nullptr,
                               sw_dgg0m1, dgg0m1b, BQ * NEDGE, 1, sbuf, gcolG, stream);
  gemm_launch<64, 128, 128, 128>(xd, nullptr, nullptr, ung2, nullptr, nullptr, sbuf, nullptr, invg,
                                 sw_dgg0m2, dgg0m2b, BQ * CITYN, 1, nd1, nullptr, stream);
  zero(sbuf, BQ * CITYN * GNNHD);
  gemm_launch<128, 1, 0, 128>(nd1, growG, nullptr, edge_w, egath, nullptr, nullptr, nullptr, nullptr,
                              sw_dgg1m1, dgg1m1b, BQ * NEDGE, 1, sbuf, gcolG, stream);
  gemm_launch<128, 128, 0, 128>(nd1, nullptr, nullptr, sbuf, nullptr, invg, nullptr, nullptr, nullptr,
                                sw_dgg1m2, dgg1m2b, BQ * CITYN, 1, nd2, nullptr, stream);

  // -------- prediction head --------
  gemm_launch<128, 0, 0, 16>(nd2, nullptr, nullptr, nullptr, nullptr, nullptr, nullptr, nullptr, nullptr,
                             sw_p1, p1b, BQ * CITYN, 1, pbuf, nullptr, stream);
  gemm_launch<16, 0, 0, 12>(pbuf, nullptr, nullptr, nullptr, nullptr, nullptr, nullptr, nullptr, nullptr,
                            sw_p2, p2b, BQ * CITYN, 0, (float*)d_out, nullptr, stream);
}
